// UnitGRU_66168266162919
// MI455X (gfx1250) — matrix operand order repaired and checked by reading, not compiled, not run on hardware
//
#include <hip/hip_runtime.h>
#include <hip/hip_bf16.h>

// ---- problem sizes ----
#define B_SZ 64
#define T_SZ 512
#define I_SZ 512
#define H_SZ 1024
#define HLAST (B_SZ * H_SZ)       // h_last elements at front of d_out

typedef __attribute__((ext_vector_type(16))) __bf16 v16bf;
typedef __attribute__((ext_vector_type(8)))  float  v8f;

__device__ __forceinline__ float sigmoidf_(float x) {
    return 1.0f / (1.0f + __expf(-x));
}

// fp32 -> bf16 bulk convert (n multiple of 1024)
__global__ __launch_bounds__(256) void gru_cvt_bf16(const float* __restrict__ src,
                                                    __bf16* __restrict__ dst, int n) {
    int i = (blockIdx.x * 256 + threadIdx.x) * 4;
    if (i >= n) return;
    float4 v = *reinterpret_cast<const float4*>(src + i);
    __bf16 tmp[4] = { (__bf16)v.x, (__bf16)v.y, (__bf16)v.z, (__bf16)v.w };
    *reinterpret_cast<uint2*>(dst + i) = *reinterpret_cast<const uint2*>(tmp);
}

// One GRU time step. Grid: (H/32, B/16). Block: 64 (= 2 wave32).
// Wave w owns the 16x16 tile at rows [blockIdx.y*16, +16),
// cols [blockIdx.x*32 + w*16, +16), fusing x_t @ Wx^T (K=512) and
// h_{t-1} @ Wh^T (K=1024) for all three gates.
// A-fragments load directly from L2-resident bf16 buffers (x precomputed,
// h from a bf16 ping-pong written by the previous step's epilogue).
__global__ __launch_bounds__(64) void gru_step(
    const __bf16* __restrict__ xbf,
    const __bf16* __restrict__ Wxr, const __bf16* __restrict__ Wxz, const __bf16* __restrict__ Wxn,
    const __bf16* __restrict__ Whr, const __bf16* __restrict__ Whz, const __bf16* __restrict__ Whn,
    const float*  __restrict__ bxr, const float*  __restrict__ bxz, const float*  __restrict__ bxn,
    const __bf16* __restrict__ hbPrev,   // bf16 h_{t-1}, [B, H]
    __bf16* __restrict__ hbNext,         // bf16 h_t,     [B, H]
    float* __restrict__ out, int t)
{
    const int wave  = threadIdx.x >> 5;
    const int lane  = threadIdx.x & 31;
    const int mBase = blockIdx.y * 16;
    const int nBase = blockIdx.x * 32 + wave * 16;

    const int rowM = mBase + (lane & 15);   // A-matrix row held by this lane
    const int nCol = nBase + (lane & 15);   // B/C/D column held by this lane
    const int kA   = (lane >> 4) * 16;      // A-fragment K offset (upper half-wave)
    const int kB   = (lane >> 4) * 16;      // B-fragment K offset (upper half-wave)

    float* __restrict__ hs = out + HLAST;   // hidden_states [B,T,H]

    // accumulators: r (with bias), z (with bias), x-part of n (with bias), h-part of n
    v8f racc, zacc, xnacc, hnacc;
    const float br = bxr[nCol], bz = bxz[nCol], bn = bxn[nCol];
    #pragma unroll
    for (int i = 0; i < 8; ++i) { racc[i] = br; zacc[i] = bz; xnacc[i] = bn; hnacc[i] = 0.0f; }

    // ---- x part: K = 512, A from precomputed bf16 inputs ----
    {
        const __bf16* xrow = xbf + ((size_t)rowM * T_SZ + t) * I_SZ;
        const __bf16* wr   = Wxr + (size_t)nCol * I_SZ;
        const __bf16* wz   = Wxz + (size_t)nCol * I_SZ;
        const __bf16* wn   = Wxn + (size_t)nCol * I_SZ;
        #pragma unroll
        for (int kk = 0; kk < I_SZ; kk += 32) {
            v16bf a;
            uint4* ap = reinterpret_cast<uint4*>(&a);
            ap[0] = *reinterpret_cast<const uint4*>(xrow + kk + kA);       // K = k0..k0+7
            ap[1] = *reinterpret_cast<const uint4*>(xrow + kk + kA + 8);   // K = k0+8..k0+15
            v16bf bwr = *reinterpret_cast<const v16bf*>(wr + kk + kB);
            v16bf bwz = *reinterpret_cast<const v16bf*>(wz + kk + kB);
            v16bf bwn = *reinterpret_cast<const v16bf*>(wn + kk + kB);
            racc  = __builtin_amdgcn_wmma_f32_16x16x32_bf16(false, a, false, bwr, (short)0, racc,  false, false);
            zacc  = __builtin_amdgcn_wmma_f32_16x16x32_bf16(false, a, false, bwz, (short)0, zacc,  false, false);
            xnacc = __builtin_amdgcn_wmma_f32_16x16x32_bf16(false, a, false, bwn, (short)0, xnacc, false, false);
        }
    }
    // ---- h part: K = 1024, A from bf16 ping-pong state ----
    if (t > 0) {
        const __bf16* hrow = hbPrev + (size_t)rowM * H_SZ;
        const __bf16* wr   = Whr + (size_t)nCol * H_SZ;
        const __bf16* wz   = Whz + (size_t)nCol * H_SZ;
        const __bf16* wn   = Whn + (size_t)nCol * H_SZ;
        #pragma unroll
        for (int kk = 0; kk < H_SZ; kk += 32) {
            v16bf a;
            uint4* ap = reinterpret_cast<uint4*>(&a);
            ap[0] = *reinterpret_cast<const uint4*>(hrow + kk + kA);
            ap[1] = *reinterpret_cast<const uint4*>(hrow + kk + kA + 8);
            v16bf bwr = *reinterpret_cast<const v16bf*>(wr + kk + kB);
            v16bf bwz = *reinterpret_cast<const v16bf*>(wz + kk + kB);
            v16bf bwn = *reinterpret_cast<const v16bf*>(wn + kk + kB);
            racc  = __builtin_amdgcn_wmma_f32_16x16x32_bf16(false, a, false, bwr, (short)0, racc,  false, false);
            zacc  = __builtin_amdgcn_wmma_f32_16x16x32_bf16(false, a, false, bwz, (short)0, zacc,  false, false);
            hnacc = __builtin_amdgcn_wmma_f32_16x16x32_bf16(false, a, false, bwn, (short)0, hnacc, false, false);
        }
    }

    // ---- epilogue: gates + state update. C/D layout: lane = col, 8 rows/lane ----
    #pragma unroll
    for (int i = 0; i < 8; ++i) {
        const int m = mBase + (lane >> 4) * 8 + i;
        float hp = 0.0f;
        if (t > 0) hp = hs[((size_t)m * T_SZ + (t - 1)) * H_SZ + nCol];   // exact fp32 h_{t-1}
        const float r  = sigmoidf_(racc[i]);
        const float z  = sigmoidf_(zacc[i]);
        const float nn = tanhf(xnacc[i] + r * hnacc[i]);
        const float hnew = (1.0f - z) * nn + z * hp;
        hs[((size_t)m * T_SZ + t) * H_SZ + nCol] = hnew;                   // fp32 output
        hbNext[(size_t)m * H_SZ + nCol] = (__bf16)hnew;                    // bf16 state for t+1
        if (t == T_SZ - 1) out[(size_t)m * H_SZ + nCol] = hnew;            // h_last
    }
}

extern "C" void kernel_launch(void* const* d_in, const int* in_sizes, int n_in,
                              void* d_out, int out_size, void* d_ws, size_t ws_size,
                              hipStream_t stream) {
    (void)in_sizes; (void)n_in; (void)out_size; (void)ws_size;
    const float* inputs = (const float*)d_in[0];
    const float* Wxr = (const float*)d_in[1];
    const float* bxr = (const float*)d_in[2];
    const float* Whr = (const float*)d_in[3];
    const float* Wxz = (const float*)d_in[4];
    const float* bxz = (const float*)d_in[5];
    const float* Whz = (const float*)d_in[6];
    const float* Wxn = (const float*)d_in[7];
    const float* bxn = (const float*)d_in[8];
    const float* Whn = (const float*)d_in[9];

    const size_t NX  = (size_t)B_SZ * T_SZ * I_SZ;  // 16,777,216
    const size_t NWX = (size_t)H_SZ * I_SZ;         //    524,288
    const size_t NWH = (size_t)H_SZ * H_SZ;         //  1,048,576
    const size_t NH  = (size_t)B_SZ * H_SZ;         //     65,536

    __bf16* xbf = (__bf16*)d_ws;
    __bf16* wxr = xbf + NX;
    __bf16* wxz = wxr + NWX;
    __bf16* wxn = wxz + NWX;
    __bf16* whr = wxn + NWX;
    __bf16* whz = whr + NWH;
    __bf16* whn = whz + NWH;
    __bf16* hb0 = whn + NWH;     // bf16 hidden-state ping-pong
    __bf16* hb1 = hb0 + NH;

    // one-time bf16 conversions (all stay L2-resident: ~43 MB << 192 MB L2)
    gru_cvt_bf16<<<dim3((unsigned)(NX  / 1024)), dim3(256), 0, stream>>>(inputs, xbf, (int)NX);
    gru_cvt_bf16<<<dim3((unsigned)(NWX / 1024)), dim3(256), 0, stream>>>(Wxr, wxr, (int)NWX);
    gru_cvt_bf16<<<dim3((unsigned)(NWX / 1024)), dim3(256), 0, stream>>>(Wxz, wxz, (int)NWX);
    gru_cvt_bf16<<<dim3((unsigned)(NWX / 1024)), dim3(256), 0, stream>>>(Wxn, wxn, (int)NWX);
    gru_cvt_bf16<<<dim3((unsigned)(NWH / 1024)), dim3(256), 0, stream>>>(Whr, whr, (int)NWH);
    gru_cvt_bf16<<<dim3((unsigned)(NWH / 1024)), dim3(256), 0, stream>>>(Whz, whz, (int)NWH);
    gru_cvt_bf16<<<dim3((unsigned)(NWH / 1024)), dim3(256), 0, stream>>>(Whn, whn, (int)NWH);

    // sequential scan: one launch per time step (launch = global barrier).
    // 128 small WGs/step -> many WGPs active on the latency-critical path.
    dim3 grid(H_SZ / 32, B_SZ / 16);   // (32, 4)
    for (int t = 0; t < T_SZ; ++t) {
        const __bf16* hp = (t & 1) ? hb1 : hb0;   // read  (unused at t=0)
        __bf16*       hn = (t & 1) ? hb0 : hb1;   // write
        gru_step<<<grid, dim3(64), 0, stream>>>(xbf, wxr, wxz, wxn, whr, whz, whn,
                                                bxr, bxz, bxn, hp, hn,
                                                (float*)d_out, t);
    }
}
